// MultiHeadAttention_88656714924288
// MI455X (gfx1250) — matrix operand order repaired and checked by reading, not compiled, not run on hardware
//
#include <hip/hip_runtime.h>
#include <hip/hip_bf16.h>

// ---------------------------------------------------------------------------
// MultiHeadAttention for MI455X (gfx1250):
//   f16 WMMA (f32 acc) + TDM tensor_load_to_lds (K tiles) +
//   global_load_async_to_lds_b128 (V tiles) + double-buffered LDS.
// N=2, S=2048, E=1024, H=16, Dh=64.
// ---------------------------------------------------------------------------

typedef _Float16 v16h __attribute__((ext_vector_type(16)));
typedef _Float16 v8h  __attribute__((ext_vector_type(8)));
typedef float    v8f  __attribute__((ext_vector_type(8)));
typedef unsigned int u32x4 __attribute__((ext_vector_type(4)));
typedef int          i32x8 __attribute__((ext_vector_type(8)));
typedef int          i32x4 __attribute__((ext_vector_type(4)));

#define S_LEN 2048
#define E_DIM 1024
#define NHEAD 16
#define DH    64

__device__ __forceinline__ v16h ld16(const _Float16* plo, const _Float16* phi) {
  v8h lo = *(const v8h*)plo;
  v8h hi = *(const v8h*)phi;
  return __builtin_shufflevector(lo, hi, 0,1,2,3,4,5,6,7,8,9,10,11,12,13,14,15);
}

__device__ __forceinline__ v8f wmma32(v16h a, v16h b, v8f c) {
  return __builtin_amdgcn_wmma_f32_16x16x32_f16(false, a, false, b, (short)0, c, false, false);
}

__device__ __forceinline__ unsigned lds_offset(const void* p) {
  return (unsigned)(unsigned long long)p;   // aperture rule: LDS addr = addr[31:0]
}

// Per-lane async copy of a 64x64 f16 tile (8KB) with 256 threads (2 x b128 each).
__device__ __forceinline__ void async_tile_64x64(const _Float16* gbase, int gstride_h,
                                                 _Float16* lbase, int tid) {
  #pragma unroll
  for (int rep = 0; rep < 2; ++rep) {
    const int c    = tid + rep * 256;     // 0..511
    const int row  = c >> 3;
    const int col8 = c & 7;
    const _Float16* g = gbase + row * gstride_h + col8 * 8;
    const unsigned l = lds_offset(lbase + row * 64 + col8 * 8);
    asm volatile("global_load_async_to_lds_b128 %0, %1, off"
                 :: "v"(l), "v"(g) : "memory");
  }
}

__device__ __forceinline__ void wait_async0() {
  asm volatile("s_wait_asynccnt 0x0" ::: "memory");
}

// TDM: one-instruction DMA of a 64x64 f16 tile into LDS (TENSORcnt).
// Descriptor layout per CDNA5 ISA (2D, no pad/iterate).
// 6-arg builtin on this toolchain: (u32x4 g0, i32x8 g1, i32x4, i32x4, i32x8, i32 cpol)
__device__ __forceinline__ void tdm_load_tile_64x64(const _Float16* g, unsigned lds_off,
                                                    unsigned stride_elems, int tid) {
#if __has_builtin(__builtin_amdgcn_tensor_load_to_lds)
  unsigned long long ga = (unsigned long long)g;
  u32x4 g0;
  g0[0] = 1u;                                             // count=1, user mode
  g0[1] = lds_off;                                        // lds_addr
  g0[2] = (unsigned)ga;                                   // global_addr[31:0]
  g0[3] = (unsigned)((ga >> 32) & 0x1FFFFFFu) | (2u << 30); // addr[56:32] | type=2
  i32x8 g1;
  g1[0] = 1 << 16;                  // wg_mask=0, data_size=1 (2 bytes)
  g1[1] = 0;                        // atomic_barrier_addr=0, tensor_dim0[15:0]=0
  g1[2] = 0x0010;                   // tensor_dim0 = 0x100000 (generous, no OOB)
  g1[3] = (64 << 16) | 0x0010;      // tensor_dim1 = 0x100000, tile_dim0 = 64
  g1[4] = 64;                       // tile_dim1 = 64, tile_dim2 = 0
  g1[5] = (int)stride_elems;        // tensor_dim0_stride[31:0]
  g1[6] = 0;                        // stride0 hi, stride1 lo
  g1[7] = 0;                        // stride1 hi
  i32x4 z4 = {0, 0, 0, 0};
  i32x8 z8 = {0, 0, 0, 0, 0, 0, 0, 0};
  __builtin_amdgcn_tensor_load_to_lds(g0, g1, z4, z4, z8, 0);
#else
  // Fallback: wave 0 (32 lanes) copies 512 chunks via async b128.
  #pragma unroll
  for (int rep = 0; rep < 16; ++rep) {
    const int c = (tid & 31) + rep * 32;
    const int row = c >> 3, col8 = c & 7;
    const _Float16* gp = g + row * stride_elems + col8 * 8;
    const unsigned l = lds_off + (unsigned)(row * 128 + col8 * 16);
    asm volatile("global_load_async_to_lds_b128 %0, %1, off"
                 :: "v"(l), "v"(gp) : "memory");
  }
#endif
}

__device__ __forceinline__ void wait_tensor0() {
#if __has_builtin(__builtin_amdgcn_tensor_load_to_lds)
  __builtin_amdgcn_s_wait_tensorcnt(0);
#else
  wait_async0();
#endif
}

// --------------------------- f32 -> f16 convert ----------------------------
__global__ __launch_bounds__(256) void cvt_f16_kernel(const float* __restrict__ in,
                                                      _Float16* __restrict__ out, int n8) {
  int i = blockIdx.x * 256 + threadIdx.x;
  if (i < n8) {
    v8f x = ((const v8f*)in)[i];
    v8h y;
    #pragma unroll
    for (int j = 0; j < 8; ++j) y[j] = (_Float16)x[j];
    ((v8h*)out)[i] = y;
  }
}

// ------------------------------- QKV GEMM ----------------------------------
// grid = dim3(32, 48): x -> 8 row-tiles of 16, y -> 64-col tile (scalar region).
// K-loop unrolled x2 with ping-pong fragment sets: no register-rotation moves.
__global__ __launch_bounds__(256) void qkv_gemm_kernel(
    const _Float16* __restrict__ xh, const _Float16* __restrict__ w,
    const float* __restrict__ bias,
    _Float16* __restrict__ qh, _Float16* __restrict__ kh, _Float16* __restrict__ vt) {
  const int lane = threadIdx.x & 31;
  const int l16  = lane & 15;
  const int hi   = lane >> 4;
  const int warp = __builtin_amdgcn_readfirstlane((int)(threadIdx.x >> 5));
  const int m0 = (blockIdx.x * 8 + warp) * 16;
  const int c0 = blockIdx.y * 64;

  v8f acc[4] = {};
  const _Float16* arow  = xh + (m0 + l16) * E_DIM;
  const _Float16* brow0 = w + (c0 + l16) * E_DIM;

  v16h a0, a1, b0[4], b1[4];
  {
    const _Float16* an = arow + hi * 8;
    a0 = ld16(an, an + 16);
    #pragma unroll
    for (int t = 0; t < 4; ++t) {
      const _Float16* bp = brow0 + t * 16 * E_DIM + hi * 8;
      b0[t] = ld16(bp, bp + 16);
    }
  }
  for (int kk = 0; kk < E_DIM; kk += 64) {
    {  // load set-1 for kk+32 while computing with set-0
      const _Float16* an = arow + (kk + 32) + hi * 8;
      __builtin_prefetch(an + 192, 0, 0);
      a1 = ld16(an, an + 16);
      #pragma unroll
      for (int t = 0; t < 4; ++t) {
        const _Float16* bp = brow0 + t * 16 * E_DIM + (kk + 32) + hi * 8;
        b1[t] = ld16(bp, bp + 16);
      }
    }
    #pragma unroll
    for (int t = 0; t < 4; ++t) acc[t] = wmma32(a0, b0[t], acc[t]);
    {  // load set-0 for kk+64 while computing with set-1
      const int k2 = (kk + 64 < E_DIM) ? (kk + 64) : kk;
      const _Float16* an = arow + k2 + hi * 8;
      a0 = ld16(an, an + 16);
      #pragma unroll
      for (int t = 0; t < 4; ++t) {
        const _Float16* bp = brow0 + t * 16 * E_DIM + k2 + hi * 8;
        b0[t] = ld16(bp, bp + 16);
      }
    }
    #pragma unroll
    for (int t = 0; t < 4; ++t) acc[t] = wmma32(a1, b1[t], acc[t]);
  }

  const int nb     = m0 >> 11;            // batch
  const int region = blockIdx.y >> 4;     // 0=q 1=k 2=v  (scalar)
  const int h      = blockIdx.y & 15;     // head         (scalar)
  #pragma unroll
  for (int t = 0; t < 4; ++t) {
    const float bv = bias[c0 + t * 16 + l16];
    const int d = t * 16 + l16;
    #pragma unroll
    for (int r = 0; r < 8; ++r) {
      const int s = (m0 + r + hi * 8) & (S_LEN - 1);
      float v = acc[t][r] + bv;
      if (region == 0) {
        qh[(((nb * NHEAD + h) * S_LEN) + s) * DH + d] = (_Float16)(v * 0.125f);
      } else if (region == 1) {
        kh[(((nb * NHEAD + h) * S_LEN) + s) * DH + d] = (_Float16)v;
      } else {
        vt[(((nb * NHEAD + h) * DH) + d) * S_LEN + s] = (_Float16)v;   // [n][h][d][s]
      }
    }
  }
}

// ---------------------------- Flash attention ------------------------------
// Block = 8 waves, one (batch, head), 128 query rows. K tile via TDM,
// V tile via per-lane async b128; double-buffered LDS; online softmax.
__global__ __launch_bounds__(256) void flash_attn_kernel(
    const _Float16* __restrict__ qh, const _Float16* __restrict__ kh,
    const _Float16* __restrict__ vt, _Float16* __restrict__ oh) {
  __shared__ __align__(16) _Float16 sK[2][64 * 64];   // [s_local][d]
  __shared__ __align__(16) _Float16 sV[2][64 * 64];   // [d][s_local]
  __shared__ __align__(16) _Float16 sP[8][16 * 64];   // per-wave P scratch

  const int tid  = threadIdx.x;
  const int warp = __builtin_amdgcn_readfirstlane(tid >> 5);
  const int lane = tid & 31;
  const int l16  = lane & 15;
  const int hi   = lane >> 4;

  const int bid    = blockIdx.x;
  const int blkrow = bid & 15;
  const int head   = (bid >> 4) & 15;
  const int nb     = bid >> 8;
  const int m0     = blkrow * 128 + warp * 16;

  const _Float16* qb = qh + ((nb * NHEAD + head) * S_LEN + m0) * DH;
  const _Float16* kb = kh + (nb * NHEAD + head) * S_LEN * DH;
  const _Float16* vb = vt + (nb * NHEAD + head) * DH * S_LEN;
  _Float16* lp = &sP[warp][0];

  v16h aq0, aq1;
  {
    const _Float16* qr = qb + l16 * DH;
    aq0 = ld16(qr + hi * 8,      qr + hi * 8 + 16);
    aq1 = ld16(qr + 32 + hi * 8, qr + 32 + hi * 8 + 16);
  }

  v8f acc[4] = {};
  float mi[8], li[8];
  #pragma unroll
  for (int r = 0; r < 8; ++r) { mi[r] = -1e30f; li[r] = 0.0f; }

  // Prologue: stage tile 0 (K via TDM from wave 0; V via async from all waves).
  if (warp == 0) tdm_load_tile_64x64(kb, lds_offset(&sK[0][0]), DH, tid);
  async_tile_64x64(vb, S_LEN, &sV[0][0], tid);
  int buf = 0;

  for (int s0 = 0; s0 < S_LEN; s0 += 64) {
    if (warp == 0) wait_tensor0();
    wait_async0();
    __syncthreads();
    if (s0 + 64 < S_LEN) {
      if (warp == 0)
        tdm_load_tile_64x64(kb + (s0 + 64) * DH, lds_offset(&sK[buf ^ 1][0]), DH, tid);
      async_tile_64x64(vb + (s0 + 64), S_LEN, &sV[buf ^ 1][0], tid);
    }
    const _Float16* kT = &sK[buf][0];
    const _Float16* vT = &sV[buf][0];

    // --- scores: 4 tiles of 16x16 over 64 keys ---
    v8f st[4];
    #pragma unroll
    for (int t = 0; t < 4; ++t) {
      const _Float16* kr = kT + (t * 16 + l16) * 64;
      v16h b0 = ld16(kr + hi * 8,      kr + hi * 8 + 16);
      v16h b1 = ld16(kr + 32 + hi * 8, kr + 32 + hi * 8 + 16);
      v8f z = {};
      z = wmma32(aq0, b0, z);
      st[t] = wmma32(aq1, b1, z);
    }

    // --- online softmax stats ---
    float mnew[8], alpha[8], rs[8];
    #pragma unroll
    for (int r = 0; r < 8; ++r) {
      float v = fmaxf(fmaxf(st[0][r], st[1][r]), fmaxf(st[2][r], st[3][r]));
      v = fmaxf(v, __shfl_xor(v, 1, 32));
      v = fmaxf(v, __shfl_xor(v, 2, 32));
      v = fmaxf(v, __shfl_xor(v, 4, 32));
      v = fmaxf(v, __shfl_xor(v, 8, 32));
      mnew[r]  = fmaxf(mi[r], v);
      alpha[r] = __expf(mi[r] - mnew[r]);
      mi[r]    = mnew[r];
      rs[r]    = 0.0f;
    }

    // --- P = exp(S - m) -> per-wave LDS scratch + row sums ---
    #pragma unroll
    for (int t = 0; t < 4; ++t) {
      #pragma unroll
      for (int r = 0; r < 8; ++r) {
        float p = __expf(st[t][r] - mnew[r]);
        rs[r] += p;
        lp[(r + hi * 8) * 64 + t * 16 + l16] = (_Float16)p;
      }
    }
    #pragma unroll
    for (int r = 0; r < 8; ++r) {
      float v = rs[r];
      v += __shfl_xor(v, 1, 32);
      v += __shfl_xor(v, 2, 32);
      v += __shfl_xor(v, 4, 32);
      v += __shfl_xor(v, 8, 32);
      li[r] = li[r] * alpha[r] + v;
      #pragma unroll
      for (int dt = 0; dt < 4; ++dt) acc[dt][r] *= alpha[r];
    }

    // --- reload P as A fragments; P @ V from LDS ---
    const _Float16* pr = lp + l16 * 64;
    v16h ap0 = ld16(pr + hi * 8,      pr + hi * 8 + 16);
    v16h ap1 = ld16(pr + 32 + hi * 8, pr + 32 + hi * 8 + 16);
    #pragma unroll
    for (int dt = 0; dt < 4; ++dt) {
      const _Float16* vr = vT + (dt * 16 + l16) * 64;
      v16h bv0 = ld16(vr + hi * 8,      vr + hi * 8 + 16);
      v16h bv1 = ld16(vr + 32 + hi * 8, vr + 32 + hi * 8 + 16);
      acc[dt] = wmma32(ap0, bv0, acc[dt]);
      acc[dt] = wmma32(ap1, bv1, acc[dt]);
    }
    buf ^= 1;
  }

  // --- normalize + write (N,S,E) f16 ---
  #pragma unroll
  for (int dt = 0; dt < 4; ++dt) {
    #pragma unroll
    for (int r = 0; r < 8; ++r) {
      const int s = m0 + r + hi * 8;
      float v = acc[dt][r] / li[r];
      oh[(nb * S_LEN + s) * E_DIM + head * DH + dt * 16 + l16] = (_Float16)v;
    }
  }
}

// --------------------------- Output projection -----------------------------
// grid = dim3(32, 16); same ping-pong pipelined K-loop.
__global__ __launch_bounds__(256) void out_proj_kernel(
    const _Float16* __restrict__ xh, const _Float16* __restrict__ w,
    const float* __restrict__ bias, float* __restrict__ out) {
  const int lane = threadIdx.x & 31;
  const int l16  = lane & 15;
  const int hi   = lane >> 4;
  const int warp = __builtin_amdgcn_readfirstlane((int)(threadIdx.x >> 5));
  const int m0 = (blockIdx.x * 8 + warp) * 16;
  const int c0 = blockIdx.y * 64;

  v8f acc[4] = {};
  const _Float16* arow  = xh + (m0 + l16) * E_DIM;
  const _Float16* brow0 = w + (c0 + l16) * E_DIM;

  v16h a0, a1, b0[4], b1[4];
  {
    const _Float16* an = arow + hi * 8;
    a0 = ld16(an, an + 16);
    #pragma unroll
    for (int t = 0; t < 4; ++t) {
      const _Float16* bp = brow0 + t * 16 * E_DIM + hi * 8;
      b0[t] = ld16(bp, bp + 16);
    }
  }
  for (int kk = 0; kk < E_DIM; kk += 64) {
    {
      const _Float16* an = arow + (kk + 32) + hi * 8;
      __builtin_prefetch(an + 192, 0, 0);
      a1 = ld16(an, an + 16);
      #pragma unroll
      for (int t = 0; t < 4; ++t) {
        const _Float16* bp = brow0 + t * 16 * E_DIM + (kk + 32) + hi * 8;
        b1[t] = ld16(bp, bp + 16);
      }
    }
    #pragma unroll
    for (int t = 0; t < 4; ++t) acc[t] = wmma32(a0, b0[t], acc[t]);
    {
      const int k2 = (kk + 64 < E_DIM) ? (kk + 64) : kk;
      const _Float16* an = arow + k2 + hi * 8;
      a0 = ld16(an, an + 16);
      #pragma unroll
      for (int t = 0; t < 4; ++t) {
        const _Float16* bp = brow0 + t * 16 * E_DIM + k2 + hi * 8;
        b0[t] = ld16(bp, bp + 16);
      }
    }
    #pragma unroll
    for (int t = 0; t < 4; ++t) acc[t] = wmma32(a1, b1[t], acc[t]);
  }
  #pragma unroll
  for (int t = 0; t < 4; ++t) {
    const float bv = bias[c0 + t * 16 + l16];
    #pragma unroll
    for (int r = 0; r < 8; ++r) {
      out[(m0 + r + hi * 8) * E_DIM + c0 + t * 16 + l16] = acc[t][r] + bv;
    }
  }
}

// ------------------------------- launcher ----------------------------------
extern "C" void kernel_launch(void* const* d_in, const int* in_sizes, int n_in,
                              void* d_out, int out_size, void* d_ws, size_t ws_size,
                              hipStream_t stream) {
  const float* query = (const float*)d_in[0];
  const float* w_in  = (const float*)d_in[3];
  const float* b_in  = (const float*)d_in[4];
  const float* w_o   = (const float*)d_in[5];
  const float* b_o   = (const float*)d_in[6];

  _Float16* ws   = (_Float16*)d_ws;
  _Float16* xh   = ws;                       // 4096*1024
  _Float16* winh = xh   + 4194304;           // 3072*1024
  _Float16* woh  = winh + 3145728;           // 1024*1024
  _Float16* qh   = woh  + 1048576;           // 2*16*2048*64
  _Float16* kh   = qh   + 4194304;
  _Float16* vth  = kh   + 4194304;           // transposed per head
  _Float16* oh   = vth  + 4194304;           // 4096*1024
  // total 25,165,824 halves = 48 MiB

  cvt_f16_kernel<<<(4194304 / 8) / 256, 256, 0, stream>>>(query, xh, 4194304 / 8);
  cvt_f16_kernel<<<(3145728 / 8) / 256, 256, 0, stream>>>(w_in, winh, 3145728 / 8);
  cvt_f16_kernel<<<(1048576 / 8) / 256, 256, 0, stream>>>(w_o, woh, 1048576 / 8);

  qkv_gemm_kernel<<<dim3(32, 48), 256, 0, stream>>>(xh, winh, b_in, qh, kh, vth);
  flash_attn_kernel<<<512, 256, 0, stream>>>(qh, kh, vth, oh);
  out_proj_kernel<<<dim3(32, 16), 256, 0, stream>>>(oh, woh, b_o, (float*)d_out);
}
